// Contextcontrast_2894807957860
// MI455X (gfx1250) — matrix operand order repaired and checked by reading, not compiled, not run on hardware
//
#include <hip/hip_runtime.h>
#include <hip/hip_bf16.h>
#include <math.h>

typedef unsigned short u16;
typedef unsigned int   u32;

typedef __attribute__((ext_vector_type(16))) __bf16 bf16x16;
typedef __attribute__((ext_vector_type(8)))  float  f32x8;

union FragAB { u32 u[8]; bf16x16 v; };
union Acc    { float f[8]; f32x8 v; };

__device__ __forceinline__ f32x8 wmma_bf16(const bf16x16 a, const bf16x16 b, const f32x8 c) {
  // D = A(16x32 bf16) * B(32x16 bf16) + C(16x16 f32)
  return __builtin_amdgcn_wmma_f32_16x16x32_bf16(false, a, false, b, (short)0, c, false, false);
}

__device__ __forceinline__ u16 f2bf(float f) {
  u32 u = __float_as_uint(f);
  u32 r = 0x7FFFu + ((u >> 16) & 1u);
  return (u16)((u + r) >> 16);
}
__device__ __forceinline__ float gelu_f(float x) {
  return 0.5f * x * (1.0f + erff(x * 0.70710678118654752440f));
}

// CDNA5 async global->LDS copy (16B per lane), tracked by ASYNCcnt.
// %0 = LDS byte offset VGPR, %1 = 64-bit global address VGPR pair.
__device__ __forceinline__ void async_ld16(u32 ldsoff, const u16* g) {
  asm volatile("global_load_async_to_lds_b128 %0, %1, off"
               :: "v"(ldsoff), "v"(g) : "memory");
}
__device__ __forceinline__ void wait_async0() {
  asm volatile("s_wait_asynccnt 0x0" ::: "memory");
}
__device__ __forceinline__ u32 lds_off(const void* p) {
  // generic LDS pointer = {aperture, offset}; low 32 bits are the LDS byte offset
  return (u32)(size_t)p;
}

// block-wide sum of (a,b) over 256 threads
__device__ __forceinline__ float2 block_sum2(float a, float b, float* s1, float* s2) {
  const int t = threadIdx.x;
  s1[t] = a; s2[t] = b;
  __syncthreads();
  for (int off = 128; off > 0; off >>= 1) {
    if (t < off) { s1[t] += s1[t + off]; s2[t] += s2[t + off]; }
    __syncthreads();
  }
  float2 r; r.x = s1[0]; r.y = s2[0];
  __syncthreads();
  return r;
}

// ---------------- elementwise converters ----------------
__global__ void k_cvt(const float* __restrict__ in, u16* __restrict__ out, int n) {
  int i = blockIdx.x * 256 + threadIdx.x;
  if (i < n) out[i] = f2bf(in[i]);
}
// in (R,C) f32 -> out (C,R) bf16
__global__ void k_cvt_t(const float* __restrict__ in, u16* __restrict__ out, int R, int C) {
  int i = blockIdx.x * 256 + threadIdx.x;
  if (i >= R * C) return;
  int r = i / C, c = i % C;
  out[(size_t)c * R + r] = f2bf(in[i]);
}
// conv2_w (OC,IC,3,3) f32 -> (tap, ic, oc) bf16
__global__ void k_cvt_conv2(const float* __restrict__ in, u16* __restrict__ out) {
  int i = blockIdx.x * 256 + threadIdx.x;
  if (i >= 9 * 1024 * 1024) return;
  int oc = i / 9216, rem = i % 9216, ic = rem / 9, tap = rem % 9;
  out[((size_t)tap * 1024 + ic) * 1024 + oc] = f2bf(in[i]);
}
// relative-position bias gather: out[(h*64+n)*64+m] = rpb[REL[n][m]*8+h]
__global__ void k_bias_tab(const float* __restrict__ rpb, float* __restrict__ out) {
  int i = blockIdx.x * 256 + threadIdx.x;
  if (i >= 8 * 64 * 64) return;
  int hh = i >> 12, n = (i >> 6) & 63, m = i & 63;
  int cn = (n >> 3) * 15 + (n & 7);
  int mf = 63 - m;
  int cm = (mf >> 3) * 15 + (mf & 7);
  out[i] = rpb[(cn + cm) * 8 + hh];
}

// ---------------- dilate + LN(norm1) + window partition -> bf16 (8192x1024) ----
__global__ __launch_bounds__(256) void k_prep_win(
    const float* __restrict__ feat, const float* __restrict__ g,
    const float* __restrict__ b, u16* __restrict__ X) {
  __shared__ float r1[256], r2[256];
  const int t = threadIdx.x, blk = blockIdx.x;
  const int bi = blk >> 8, hd = (blk >> 4) & 15, wd = blk & 15;
  // dilated(hd,wd) <- feat[(hd&7)*2+(hd>>3), (wd&7)*2+(wd>>3)]
  const int h = (hd & 7) * 2 + (hd >> 3), w = (wd & 7) * 2 + (wd >> 3);
  // window row: win=(hd>>3, wd>>3), token=(hd&7, wd&7)
  const int row = ((bi * 2 + (hd >> 3)) * 2 + (wd >> 3)) * 64 + (hd & 7) * 8 + (wd & 7);
  const float* src = feat + (size_t)bi * 262144 + h * 16 + w;
  float x[4], s = 0.f, q = 0.f;
#pragma unroll
  for (int i = 0; i < 4; ++i) {
    const int c = t + i * 256;
    x[i] = src[(size_t)c * 256];
    s += x[i]; q += x[i] * x[i];
  }
  float2 r = block_sum2(s, q, r1, r2);
  const float mean = r.x * (1.0f / 1024.0f);
  const float inv = rsqrtf(r.y * (1.0f / 1024.0f) - mean * mean + 1e-5f);
#pragma unroll
  for (int i = 0; i < 4; ++i) {
    const int c = t + i * 256;
    X[(size_t)row * 1024 + c] = f2bf((x[i] - mean) * inv * g[c] + b[c]);
  }
}

// ------- generic WMMA GEMM: D = A(MxK bf16)*B(KxN bf16); async double-buffer --
#define GM_BM 128
#define GM_BN 64
#define GM_BK 32
__global__ __launch_bounds__(256) void k_gemm(
    const u16* __restrict__ A, const u16* __restrict__ B,
    const float* __restrict__ bias, float* __restrict__ outF,
    u16* __restrict__ outH, int M, int N, int K, float scale, int act) {
  __shared__ __align__(16) u16 sA[2][GM_BM * GM_BK];
  __shared__ __align__(16) u16 sB[2][GM_BK * GM_BN];
  const int t = threadIdx.x, lane = t & 31, wv = t >> 5;
  const int m0 = blockIdx.x * GM_BM, n0 = blockIdx.y * GM_BN;
  const int wm = (wv >> 1) * 32, wn = (wv & 1) * 32;
  const int hi = lane >> 4, ml = lane & 15;
  Acc acc[2][2];
#pragma unroll
  for (int i = 0; i < 2; ++i)
#pragma unroll
    for (int j = 0; j < 2; ++j)
#pragma unroll
      for (int r = 0; r < 8; ++r) acc[i][j].f[r] = 0.0f;

  // staging coordinates
  const int ra = t >> 1, ca = (t & 1) * 16;  // A: 128 rows x 32, 16 halves/thread
  const int rb = t >> 3, cb = (t & 7) * 8;   // B: 32 rows x 64, 8 halves/thread
  const u16* gA = A + (size_t)(m0 + ra) * K + ca;
  const u16* gB = B + (size_t)rb * N + n0 + cb;
  const u32 la0 = lds_off(&sA[0][ra * GM_BK + ca]);
  const u32 la1 = lds_off(&sA[1][ra * GM_BK + ca]);
  const u32 lb0 = lds_off(&sB[0][rb * GM_BN + cb]);
  const u32 lb1 = lds_off(&sB[1][rb * GM_BN + cb]);

  // prologue: stage 0
  async_ld16(la0, gA);
  async_ld16(la0 + 16, gA + 8);
  async_ld16(lb0, gB);

  const int nk = K / GM_BK;
  for (int i = 0; i < nk; ++i) {
    const int cur = i & 1;
    wait_async0();          // stage i resident (and any stage i-1 prefetch drained)
    __syncthreads();        // visible to all waves; all waves done reading buf cur^1
    if (i + 1 < nk) {       // prefetch stage i+1 into the other buffer (overlaps WMMAs)
      const u16* gA2 = gA + (i + 1) * GM_BK;
      const u16* gB2 = gB + (size_t)(i + 1) * GM_BK * N;
      async_ld16(cur ? la0 : la1, gA2);
      async_ld16((cur ? la0 : la1) + 16, gA2 + 8);
      async_ld16(cur ? lb0 : lb1, gB2);
    }
    FragAB fa[2], fb[2];
#pragma unroll
    for (int ii = 0; ii < 2; ++ii) {
      const u16* base = &sA[cur][(wm + ii * 16 + ml) * GM_BK];
#pragma unroll
      for (int jj = 0; jj < 8; ++jj) {
        const int kk = ((jj < 4) ? (jj * 2) : (jj * 2 + 8)) + hi * 8;
        fa[ii].u[jj] = *(const u32*)(base + kk);
      }
    }
#pragma unroll
    for (int j = 0; j < 2; ++j) {
      const u16* base = &sB[cur][wn + j * 16 + ml];
#pragma unroll
      for (int jj = 0; jj < 8; ++jj) {
        const int kk = ((jj < 4) ? (jj * 2) : (jj * 2 + 8)) + hi * 8;
        fb[j].u[jj] = (u32)base[kk * GM_BN] | ((u32)base[(kk + 1) * GM_BN] << 16);
      }
    }
#pragma unroll
    for (int ii = 0; ii < 2; ++ii)
#pragma unroll
      for (int j = 0; j < 2; ++j)
        acc[ii][j].v = wmma_bf16(fa[ii].v, fb[j].v, acc[ii][j].v);
    __syncthreads();        // done reading buf cur before it is refilled at i+2
  }
  const int nl = lane & 15;
#pragma unroll
  for (int i = 0; i < 2; ++i)
#pragma unroll
    for (int j = 0; j < 2; ++j) {
      const int col = n0 + wn + j * 16 + nl;
      const float bb = bias ? bias[col] : 0.0f;
#pragma unroll
      for (int r = 0; r < 8; ++r) {
        const int row = m0 + wm + i * 16 + hi * 8 + r;
        float v = (acc[i][j].f[r] + bb) * scale;
        if (act) v = gelu_f(v);
        const size_t o = (size_t)row * N + col;
        if (outF) outF[o] = v;
        if (outH) outH[o] = f2bf(v);
      }
    }
}

// ---------------- fused window attention: per (window, head) ------------------
__global__ __launch_bounds__(256) void k_attn(
    const u16* __restrict__ Q, const u16* __restrict__ Kq, const u16* __restrict__ V,
    const float* __restrict__ biasTab, u16* __restrict__ O) {
  __shared__ __align__(16) unsigned char smem[65536];
  u16*   sQ  = (u16*)smem;              // 64 x 128 row-major
  u16*   sKT = (u16*)(smem + 16384);    // 128 x 64 (d-major, i.e. K transposed)
  u16*   sV  = (u16*)(smem + 32768);    // 64 x 128 row-major
  float* sS  = (float*)(smem + 49152);  // 64 x 64 scores
  u16*   sP  = (u16*)smem;              // aliases sQ (dead after score phase)

  const int t = threadIdx.x, lane = t & 31, wv = t >> 5;
  const int win = blockIdx.x, hh = blockIdx.y;
  const size_t rowbase = (size_t)win * 64;
  const int colbase = hh * 128;
  const int hi = lane >> 4, ml = lane & 15;

  {  // stage Q, V row-major; K transposed
    const int r = t >> 2, seg = (t & 3) * 32;
    const u16* gq = Q  + (rowbase + r) * 1024 + colbase + seg;
    const u16* gk = Kq + (rowbase + r) * 1024 + colbase + seg;
    const u16* gv = V  + (rowbase + r) * 1024 + colbase + seg;
#pragma unroll
    for (int i = 0; i < 4; ++i) {
      ((uint4*)&sQ[r * 128 + seg])[i] = ((const uint4*)gq)[i];
      ((uint4*)&sV[r * 128 + seg])[i] = ((const uint4*)gv)[i];
    }
#pragma unroll
    for (int i = 0; i < 32; ++i) sKT[(seg + i) * 64 + r] = gk[i];
  }
  __syncthreads();

  {  // S = Q*K^T + bias ; wave -> 16x32 output strip
    const int tm = wv >> 1;
    Acc acc[2];
#pragma unroll
    for (int j = 0; j < 2; ++j)
#pragma unroll
      for (int r = 0; r < 8; ++r) acc[j].f[r] = 0.0f;
#pragma unroll
    for (int ks = 0; ks < 4; ++ks) {
      FragAB fa;
      const u16* base = &sQ[(tm * 16 + ml) * 128 + ks * 32];
#pragma unroll
      for (int jj = 0; jj < 8; ++jj) {
        const int kk = ((jj < 4) ? jj * 2 : jj * 2 + 8) + hi * 8;
        fa.u[jj] = *(const u32*)(base + kk);
      }
#pragma unroll
      for (int j = 0; j < 2; ++j) {
        const int tn = (wv & 1) * 2 + j;
        FragAB fb;
        const u16* bb = &sKT[(ks * 32) * 64 + tn * 16 + ml];
#pragma unroll
        for (int jj = 0; jj < 8; ++jj) {
          const int kk = ((jj < 4) ? jj * 2 : jj * 2 + 8) + hi * 8;
          fb.u[jj] = (u32)bb[kk * 64] | ((u32)bb[(kk + 1) * 64] << 16);
        }
        acc[j].v = wmma_bf16(fa.v, fb.v, acc[j].v);
      }
    }
    const float* bt = biasTab + hh * 4096;
    const int nl = lane & 15;
#pragma unroll
    for (int j = 0; j < 2; ++j) {
      const int tn = (wv & 1) * 2 + j;
#pragma unroll
      for (int r = 0; r < 8; ++r) {
        const int m = tm * 16 + hi * 8 + r, n = tn * 16 + nl;
        sS[m * 64 + n] = acc[j].f[r] + bt[m * 64 + n];
      }
    }
  }
  __syncthreads();

  if (t < 64) {  // f32 softmax over row t
    float mx = -3.0e38f;
    for (int n = 0; n < 64; ++n) mx = fmaxf(mx, sS[t * 64 + n]);
    float s = 0.0f;
    for (int n = 0; n < 64; ++n) {
      const float e = __expf(sS[t * 64 + n] - mx);
      sS[t * 64 + n] = e;
      s += e;
    }
    const float inv = 1.0f / s;
    for (int n = 0; n < 64; ++n) sP[t * 64 + n] = f2bf(sS[t * 64 + n] * inv);
  }
  __syncthreads();

  {  // O = P*V ; wave -> 16x64 output strip
    const int tm = wv >> 1, tnb = (wv & 1) * 4;
    Acc acc[4];
#pragma unroll
    for (int j = 0; j < 4; ++j)
#pragma unroll
      for (int r = 0; r < 8; ++r) acc[j].f[r] = 0.0f;
#pragma unroll
    for (int ks = 0; ks < 2; ++ks) {
      FragAB fa;
      const u16* base = &sP[(tm * 16 + ml) * 64 + ks * 32];
#pragma unroll
      for (int jj = 0; jj < 8; ++jj) {
        const int kk = ((jj < 4) ? jj * 2 : jj * 2 + 8) + hi * 8;
        fa.u[jj] = *(const u32*)(base + kk);
      }
#pragma unroll
      for (int j = 0; j < 4; ++j) {
        FragAB fb;
        const u16* bb = &sV[(ks * 32) * 128 + (tnb + j) * 16 + ml];
#pragma unroll
        for (int jj = 0; jj < 8; ++jj) {
          const int kk = ((jj < 4) ? jj * 2 : jj * 2 + 8) + hi * 8;
          fb.u[jj] = (u32)bb[kk * 128] | ((u32)bb[(kk + 1) * 128] << 16);
        }
        acc[j].v = wmma_bf16(fa.v, fb.v, acc[j].v);
      }
    }
    const int nl = lane & 15;
#pragma unroll
    for (int j = 0; j < 4; ++j)
#pragma unroll
      for (int r = 0; r < 8; ++r) {
        const int m = tm * 16 + hi * 8 + r;
        const int col = colbase + (tnb + j) * 16 + nl;
        O[(rowbase + m) * 1024 + col] = f2bf(acc[j].f[r]);
      }
  }
}

// ----- ctx = feat + unwindow/undilate(proj) ; LN(norm2) -> bf16 for fc1 -------
__global__ __launch_bounds__(256) void k_ctx_ln(
    const float* __restrict__ feat, const float* __restrict__ oproj,
    const float* __restrict__ g, const float* __restrict__ b,
    float* __restrict__ ctxRow, u16* __restrict__ lnt) {
  __shared__ float r1[256], r2[256];
  const int t = threadIdx.x, blk = blockIdx.x;
  const int bi = blk >> 8, hw = blk & 255, h = hw >> 4, w = hw & 15;
  const int wi = h & 1, si = h >> 1, wj = w & 1, sj = w >> 1;
  const int arow = ((bi * 2 + wi) * 2 + wj) * 64 + si * 8 + sj;
  float x[4], s = 0.f, q = 0.f;
#pragma unroll
  for (int i = 0; i < 4; ++i) {
    const int c = t + i * 256;
    const float v = feat[((size_t)bi * 1024 + c) * 256 + hw] + oproj[(size_t)arow * 1024 + c];
    ctxRow[(size_t)blk * 1024 + c] = v;
    x[i] = v; s += v; q += v * v;
  }
  float2 r = block_sum2(s, q, r1, r2);
  const float mean = r.x * (1.0f / 1024.0f);
  const float inv = rsqrtf(r.y * (1.0f / 1024.0f) - mean * mean + 1e-5f);
#pragma unroll
  for (int i = 0; i < 4; ++i) {
    const int c = t + i * 256;
    lnt[(size_t)blk * 1024 + c] = f2bf((x[i] - mean) * inv * g[c] + b[c]);
  }
}

// ----- t+=mlp; cc=gelu(LN3(local-ctx)); build [local|ctx|cc] bf16 (8192x3072) --
__global__ __launch_bounds__(256) void k_fuse_conv_in(
    const float* __restrict__ localx, const float* __restrict__ ctxRow,
    const float* __restrict__ hout, const float* __restrict__ g,
    const float* __restrict__ b, u16* __restrict__ Ain) {
  __shared__ float r1[256], r2[256];
  const int t = threadIdx.x, blk = blockIdx.x;
  const int bi = blk >> 8, hw = blk & 255;
  float lv[4], tn[4], d[4], s = 0.f, q = 0.f;
#pragma unroll
  for (int i = 0; i < 4; ++i) {
    const int c = t + i * 256;
    tn[i] = ctxRow[(size_t)blk * 1024 + c] + hout[(size_t)blk * 1024 + c];
    lv[i] = localx[((size_t)bi * 1024 + c) * 256 + hw];
    d[i] = lv[i] - tn[i];
    s += d[i]; q += d[i] * d[i];
  }
  float2 r = block_sum2(s, q, r1, r2);
  const float mean = r.x * (1.0f / 1024.0f);
  const float inv = rsqrtf(r.y * (1.0f / 1024.0f) - mean * mean + 1e-5f);
  const size_t base = (size_t)blk * 3072;
#pragma unroll
  for (int i = 0; i < 4; ++i) {
    const int c = t + i * 256;
    const float cc = gelu_f((d[i] - mean) * inv * g[c] + b[c]);
    Ain[base + c]        = f2bf(lv[i]);
    Ain[base + 1024 + c] = f2bf(tn[i]);
    Ain[base + 2048 + c] = f2bf(cc);
  }
}

// ---------------- 3x3 conv as implicit GEMM (9 taps x 32-deep K chunks) -------
__global__ __launch_bounds__(256) void k_conv3x3(
    const u16* __restrict__ X, const u16* __restrict__ W,
    const float* __restrict__ bias, float* __restrict__ out) {
  __shared__ __align__(16) u16 sA[64 * 32];
  __shared__ __align__(16) u16 sB[32 * 64];
  const int t = threadIdx.x, lane = t & 31, wv = t >> 5;
  const int m0 = blockIdx.x * 64, n0 = blockIdx.y * 64;
  const int wm = (wv >> 1) * 16, wn = (wv & 1) * 32;
  const int hi = lane >> 4, ml = lane & 15;
  Acc acc[2];
#pragma unroll
  for (int j = 0; j < 2; ++j)
#pragma unroll
    for (int r = 0; r < 8; ++r) acc[j].f[r] = 0.0f;

  for (int tap = 0; tap < 9; ++tap) {
    const int dy = tap / 3 - 1, dx = tap % 3 - 1;
    for (int k0 = 0; k0 < 1024; k0 += 32) {
      {
        const int r = t >> 2, c = (t & 3) * 8;
        const int p = m0 + r;
        const int b = p >> 8, h = ((p >> 4) & 15) + dy, w = (p & 15) + dx;
        uint4 val; val.x = val.y = val.z = val.w = 0u;
        if ((unsigned)h < 16u && (unsigned)w < 16u)
          val = *(const uint4*)(X + ((size_t)(b * 256 + h * 16 + w)) * 1024 + k0 + c);
        *(uint4*)&sA[r * 32 + c] = val;
      }
      {
        const int r = t >> 3, c = (t & 7) * 8;
        *(uint4*)&sB[r * 64 + c] =
            *(const uint4*)(W + ((size_t)(tap * 1024 + k0 + r)) * 1024 + n0 + c);
      }
      __syncthreads();
      FragAB fa;
#pragma unroll
      for (int jj = 0; jj < 8; ++jj) {
        const int kk = ((jj < 4) ? jj * 2 : jj * 2 + 8) + hi * 8;
        fa.u[jj] = *(const u32*)(&sA[(wm + ml) * 32] + kk);
      }
#pragma unroll
      for (int j = 0; j < 2; ++j) {
        FragAB fb;
        const u16* base = &sB[wn + j * 16 + ml];
#pragma unroll
        for (int jj = 0; jj < 8; ++jj) {
          const int kk = ((jj < 4) ? jj * 2 : jj * 2 + 8) + hi * 8;
          fb.u[jj] = (u32)base[kk * 64] | ((u32)base[(kk + 1) * 64] << 16);
        }
        acc[j].v = wmma_bf16(fa.v, fb.v, acc[j].v);
      }
      __syncthreads();
    }
  }
  const int nl = lane & 15;
#pragma unroll
  for (int j = 0; j < 2; ++j) {
    const int col = n0 + wn + j * 16 + nl;
    const float bb = bias[col];
#pragma unroll
    for (int r = 0; r < 8; ++r) {
      const int row = m0 + wm + hi * 8 + r;
      out[(size_t)row * 1024 + col] = acc[j].f[r] + bb;
    }
  }
}

// ---------------- BN stats (train-mode, biased var) + BN/ReLU epilogue --------
__global__ __launch_bounds__(256) void k_bn_stats(
    const float* __restrict__ Y, float* __restrict__ mean, float* __restrict__ inv) {
  __shared__ float r1[256], r2[256];
  const int c = blockIdx.x, t = threadIdx.x;
  float s = 0.f, q = 0.f;
  for (int r = t; r < 8192; r += 256) {
    const float v = Y[(size_t)r * 1024 + c];
    s += v; q += v * v;
  }
  float2 rr = block_sum2(s, q, r1, r2);
  if (t == 0) {
    const float m = rr.x * (1.0f / 8192.0f);
    mean[c] = m;
    inv[c] = rsqrtf(rr.y * (1.0f / 8192.0f) - m * m + 1e-5f);
  }
}
__global__ void k_bn_out(const float* __restrict__ Y, const float* __restrict__ mean,
                         const float* __restrict__ inv, const float* __restrict__ g,
                         const float* __restrict__ b, float* __restrict__ out) {
  int i = blockIdx.x * 256 + threadIdx.x;
  if (i >= 32 * 1024 * 256) return;
  const int bi = i >> 18, c = (i >> 8) & 1023, hw = i & 255;
  const float v = Y[((size_t)(bi * 256 + hw)) * 1024 + c];
  const float y = (v - mean[c]) * inv[c] * g[c] + b[c];
  out[i] = fmaxf(y, 0.0f);
}

// -----------------------------------------------------------------------------
extern "C" void kernel_launch(void* const* d_in, const int* in_sizes, int n_in,
                              void* d_out, int out_size, void* d_ws, size_t ws_size,
                              hipStream_t stream) {
  (void)in_sizes; (void)n_in; (void)out_size; (void)ws_size;
  const float* feat    = (const float*)d_in[0];
  const float* localx  = (const float*)d_in[1];
  const float* norm1_w = (const float*)d_in[2];
  const float* norm1_b = (const float*)d_in[3];
  const float* rpb     = (const float*)d_in[4];
  const float* q_w = (const float*)d_in[5];  const float* q_b = (const float*)d_in[6];
  const float* k_w = (const float*)d_in[7];  const float* k_b = (const float*)d_in[8];
  const float* v_w = (const float*)d_in[9];  const float* v_b = (const float*)d_in[10];
  const float* p_w = (const float*)d_in[11]; const float* p_b = (const float*)d_in[12];
  const float* n2w = (const float*)d_in[13]; const float* n2b = (const float*)d_in[14];
  const float* fc1w = (const float*)d_in[15]; const float* fc1b = (const float*)d_in[16];
  const float* fc2w = (const float*)d_in[17]; const float* fc2b = (const float*)d_in[18];
  const float* n3w = (const float*)d_in[19]; const float* n3b = (const float*)d_in[20];
  const float* c1w = (const float*)d_in[21]; const float* c1b = (const float*)d_in[22];
  const float* c2w = (const float*)d_in[23]; const float* c2b = (const float*)d_in[24];
  const float* bnw = (const float*)d_in[25]; const float* bnb = (const float*)d_in[26];

  char* wsp = (char*)d_ws;
  size_t off = 0;
  auto alloc = [&](size_t bytes) -> void* {
    void* p = wsp + off;
    off += (bytes + 255) & ~(size_t)255;
    return p;
  };
  // weights (bf16) + small tables
  u16* WQ   = (u16*)alloc(2097152);
  u16* WK   = (u16*)alloc(2097152);
  u16* WV   = (u16*)alloc(2097152);
  u16* WP   = (u16*)alloc(2097152);
  u16* WFC1 = (u16*)alloc(4194304);
  u16* WFC2 = (u16*)alloc(4194304);
  u16* WC1T = (u16*)alloc(6291456);
  u16* WC2  = (u16*)alloc(18874368);
  float* BIAS = (float*)alloc(131072);
  float* BNM  = (float*)alloc(4096);
  float* BNI  = (float*)alloc(4096);
  // activations (aliased across phases)
  u16*   XWIN  = (u16*)alloc(16777216);   // reused as OB after QKV GEMMs
  u16*   QB    = (u16*)alloc(16777216);   // reused as LNT after attention
  u16*   KB    = (u16*)alloc(16777216);   // reused as Y1 after attention
  u16*   VB    = (u16*)alloc(16777216);
  float* OPROJ = (float*)alloc(33554432); // reused as HOUT after k_ctx_ln
  float* CTXR  = (float*)alloc(33554432); // reused as Y2 after k_fuse_conv_in
  u16*   HMID  = (u16*)alloc(33554432);   // 8192 x 2048
  u16*   ACONV = (u16*)alloc(50331648);   // 8192 x 3072
  u16*   OB   = XWIN;
  u16*   LNT  = QB;
  u16*   Y1   = KB;
  float* HOUT = OPROJ;
  float* Y2   = CTXR;

  const dim3 blk(256);
  const int M = 8192;
  const float qscale = 0.08838834764831845f;  // 128^-0.5

  // weight conversion
  k_cvt<<<4096, blk, 0, stream>>>(q_w, WQ, 1048576);
  k_cvt<<<4096, blk, 0, stream>>>(k_w, WK, 1048576);
  k_cvt<<<4096, blk, 0, stream>>>(v_w, WV, 1048576);
  k_cvt<<<4096, blk, 0, stream>>>(p_w, WP, 1048576);
  k_cvt<<<8192, blk, 0, stream>>>(fc1w, WFC1, 2097152);
  k_cvt<<<8192, blk, 0, stream>>>(fc2w, WFC2, 2097152);
  k_cvt_t<<<12288, blk, 0, stream>>>(c1w, WC1T, 1024, 3072);
  k_cvt_conv2<<<36864, blk, 0, stream>>>(c2w, WC2);
  k_bias_tab<<<128, blk, 0, stream>>>(rpb, BIAS);

  // dilate + LN + window partition
  k_prep_win<<<8192, blk, 0, stream>>>(feat, norm1_w, norm1_b, XWIN);

  // QKV projections (Q folded with scale)
  const dim3 g16(64, 16), g32(64, 32);
  k_gemm<<<g16, blk, 0, stream>>>(XWIN, WQ, q_b, nullptr, QB, M, 1024, 1024, qscale, 0);
  k_gemm<<<g16, blk, 0, stream>>>(XWIN, WK, k_b, nullptr, KB, M, 1024, 1024, 1.0f, 0);
  k_gemm<<<g16, blk, 0, stream>>>(XWIN, WV, v_b, nullptr, VB, M, 1024, 1024, 1.0f, 0);

  // fused attention per (window, head)
  k_attn<<<dim3(128, 8), blk, 0, stream>>>(QB, KB, VB, BIAS, OB);

  // output projection, residual + LN2
  k_gemm<<<g16, blk, 0, stream>>>(OB, WP, p_b, OPROJ, nullptr, M, 1024, 1024, 1.0f, 0);
  k_ctx_ln<<<8192, blk, 0, stream>>>(feat, OPROJ, n2w, n2b, CTXR, LNT);

  // MLP
  k_gemm<<<g32, blk, 0, stream>>>(LNT, WFC1, fc1b, nullptr, HMID, M, 2048, 1024, 1.0f, 1);
  k_gemm<<<g16, blk, 0, stream>>>(HMID, WFC2, fc2b, HOUT, nullptr, M, 1024, 2048, 1.0f, 0);

  // residual + contrast branch + concat for 1x1 conv
  k_fuse_conv_in<<<8192, blk, 0, stream>>>(localx, CTXR, HOUT, n3w, n3b, ACONV);

  // 1x1 conv (GEMM over 3C) then 3x3 conv (implicit GEMM)
  k_gemm<<<g16, blk, 0, stream>>>(ACONV, WC1T, c1b, nullptr, Y1, M, 1024, 3072, 1.0f, 0);
  k_conv3x3<<<dim3(128, 16), blk, 0, stream>>>(Y1, WC2, c2b, Y2);

  // batch norm (train mode) + relu into NCHW output
  k_bn_stats<<<1024, blk, 0, stream>>>(Y2, BNM, BNI);
  k_bn_out<<<32768, blk, 0, stream>>>(Y2, BNM, BNI, bnw, bnb, (float*)d_out);
}
